// RoIHeads_12919261626793
// MI455X (gfx1250) — compile-verified, matrix operand order checked by reading, not run on hardware
//
#include <hip/hip_runtime.h>
#include <hip/hip_bf16.h>

// ---------------- constants from the reference ----------------
#define NPROP 100000
#define NCLS  91
#define NFG   90          // foreground classes 1..90
#define TOPK  1000
#define NDETS 100
#define CAP   4096        // per-class candidate cap before sort (s > 0.05 compaction)

__device__ __constant__ float kImgW = 1333.0f;
__device__ __constant__ float kImgH = 800.0f;
#define SCORE_THRESH 0.05f
#define NMS_THRESH   0.5f
#define BBOX_CLIP    4.135166556742356f   // log(1000/16)

// ---------------- CDNA5 types ----------------
typedef __attribute__((ext_vector_type(16))) _Float16 v16h;
typedef __attribute__((ext_vector_type(8)))  float    v8f;
typedef unsigned int u32x4 __attribute__((ext_vector_type(4)));
typedef int          i32x4 __attribute__((ext_vector_type(4)));
typedef int          i32x8 __attribute__((ext_vector_type(8)));

// TDM 1-D bulk copy global->LDS (descriptor per CDNA5 ISA).
// ndw = number of dwords (data_size = 4B). Waits on TENSORcnt before returning.
__device__ __forceinline__ void tdm_load_1d(unsigned lds_off, const void* gptr, unsigned ndw) {
#if defined(__gfx1250__)
  unsigned long long ga = (unsigned long long)gptr;
  u32x4 g0;
  g0[0] = 1u;                                   // count=1, user descriptor
  g0[1] = lds_off;                              // lds_addr
  g0[2] = (unsigned)ga;                         // global_addr[31:0]
  g0[3] = (unsigned)((ga >> 32) & 0x01FFFFFFu) | (2u << 30);  // addr[56:32] | type=2
  i32x8 g1;
  g1[0] = (int)(2u << 16);                      // workgroup_mask=0, data_size=2 (4B)
  g1[1] = (int)((ndw & 0xFFFFu) << 16);         // tensor_dim0[15:0] in [31:16]
  g1[2] = (int)(((ndw >> 16) & 0xFFFFu) | (1u << 16)); // tensor_dim0 hi | tensor_dim1=1 lo
  g1[3] = (int)((ndw & 0xFFFFu) << 16);         // tensor_dim1 hi=0 | tile_dim0
  g1[4] = 1;                                    // tile_dim1=1, tile_dim2=0
  g1[5] = (int)ndw;                             // tensor_dim0_stride lo
  g1[6] = 0;
  g1[7] = 0;
  i32x4 z4 = {};
#if defined(__clang_major__) && (__clang_major__ >= 23)
  i32x8 z8 = {};
  __builtin_amdgcn_tensor_load_to_lds(g0, g1, z4, z4, z8, 0);
#else
  __builtin_amdgcn_tensor_load_to_lds(g0, g1, z4, z4, 0);
#endif
  __builtin_amdgcn_s_wait_tensorcnt(0);
#else
  (void)lds_off; (void)gptr; (void)ndw;
#endif
}

// =====================================================================
// Stage A: fused softmax -> class-major scores.
// One wave handles 16 proposals. The wave's 16x91-float logit block is
// staged through LDS with fully-coalesced flat loads; row sums
// Sum_k exp(x-max) are computed with V_WMMA_F32_16X16X32_F16
// (A = exp tile f16, B = ones, K padded 91->96 with compile-time zeros).
// The WMMA (XDL, TRANS-tracked) co-executes with the VALU/TRANS exp stream.
// =====================================================================
#define ROWELEMS (16 * NCLS)   // 1456 floats per wave tile

__global__ __launch_bounds__(256) void softmax_scores_kernel(
    const float* __restrict__ logits, float* __restrict__ scores_t) {
  __shared__ float lbuf[8][ROWELEMS];   // 8 waves x 5824 B = 46.6 KB
  __shared__ float s_sum[8][16];
  const int wave = threadIdx.x >> 5;
  const int lane = threadIdx.x & 31;
  const int gwave = blockIdx.x * 8 + wave;
  const int p0 = gwave * 16;
  const bool active = (p0 < NPROP);     // uniform per wave (NPROP % 16 == 0)

  const int m = lane & 15;   // A-matrix row carried by this lane
  const int h = lane >> 4;   // lane-half selects which K's this lane holds
  const int p = p0 + m;

  // ---- coalesced global -> LDS stage (lane-contiguous dwords) ----
  if (active) {
    const float* __restrict__ src = logits + (size_t)p0 * NCLS;
    __builtin_prefetch(src + 8 * ROWELEMS, 0, 1);  // global_prefetch_b8: next block's tile
    for (int f = lane; f < ROWELEMS; f += 32) lbuf[wave][f] = src[f];
  }
  __syncthreads();

  float es[3][16];           // exp(x - max), reused for fragment + store
  if (active) {
    const float* __restrict__ rb = &lbuf[wave][m * NCLS];
    float mx = -INFINITY;
    float xs[3][16];
#pragma unroll
    for (int t = 0; t < 3; ++t) {
#pragma unroll
      for (int i = 0; i < 16; ++i) {
        // 16-bit A layout: lane-half h holds K in {8h..8h+7} and {16+8h..23+8h} per 32-K tile
        const int k = 32 * t + (i < 8 ? (8 * h + i) : (16 + 8 * h + (i - 8)));
        if (k < NCLS) {                       // compile-time: folds away for pad slots
          float v = rb[k];
          xs[t][i] = v;
          mx = fmaxf(mx, v);
        } else {
          xs[t][i] = 0.0f;
        }
      }
    }
    mx = fmaxf(mx, __shfl_xor(mx, 16, 32));   // combine the two lane-halves of each row

    v16h b;
#pragma unroll
    for (int i = 0; i < 16; ++i) b[i] = (_Float16)1.0f;
    v8f acc = {};
#pragma unroll
    for (int t = 0; t < 3; ++t) {
      v16h a;
#pragma unroll
      for (int i = 0; i < 16; ++i) {
        const int k = 32 * t + (i < 8 ? (8 * h + i) : (16 + 8 * h + (i - 8)));
        if (k < NCLS) {
          float e = __expf(xs[t][i] - mx);    // single exp; f32 kept for store
          es[t][i] = e;
          a[i] = (_Float16)e;
        } else {
          es[t][i] = 0.0f;
          a[i] = (_Float16)0.0f;              // compile-time zero padding
        }
      }
      acc = __builtin_amdgcn_wmma_f32_16x16x32_f16(
          /*neg_a=*/false, a, /*neg_b=*/false, b,
          /*c_mod=*/(short)0, acc, /*reuse_a=*/false, /*reuse_b=*/false);
    }
    // D layout: VGPR r -> M=r (lanes 0-15) / M=8+r (lanes 16-31); every column equal.
    if (m == 0) {
#pragma unroll
      for (int r = 0; r < 8; ++r) s_sum[wave][h * 8 + r] = acc[r];
    }
  }
  __syncthreads();
  if (active) {
    const float inv = 1.0f / s_sum[wave][m];
#pragma unroll
    for (int t = 0; t < 3; ++t) {
#pragma unroll
      for (int i = 0; i < 16; ++i) {
        const int k = 32 * t + (i < 8 ? (8 * h + i) : (16 + 8 * h + (i - 8)));
        if (k >= 1 && k < NCLS) {
          // class-major store: lanes 0..15 of a half write 64B contiguous
          scores_t[(size_t)(k - 1) * NPROP + p] = es[t][i] * inv;
        }
      }
    }
  }
}

// =====================================================================
// Stage B1: per-class compaction of candidates with s > 0.05
// =====================================================================
__global__ void zero_cnt_kernel(int* cnt) {
  int i = blockIdx.x * blockDim.x + threadIdx.x;
  if (i < NFG) cnt[i] = 0;
}

__global__ __launch_bounds__(256) void compact_kernel(
    const float* __restrict__ scores_t, int* __restrict__ cnt,
    unsigned long long* __restrict__ cand) {
  int c = blockIdx.y;
  int p = blockIdx.x * blockDim.x + threadIdx.x;
  if (p >= NPROP) return;
  float s = scores_t[(size_t)c * NPROP + p];
  if (s > SCORE_THRESH) {
    int pos = atomicAdd(&cnt[c], 1);
    if (pos < CAP) {
      unsigned u = __float_as_uint(s) | 0x80000000u;  // order-preserving key (s > 0)
      cand[(size_t)c * CAP + pos] = ((unsigned long long)u << 32) | (unsigned)p;
    }
  }
}

// =====================================================================
// Stage B2: TDM-stage candidate buffer into LDS, bitonic sort descending,
// emit top-1000 (index, score) per class.
// =====================================================================
__global__ __launch_bounds__(1024) void sort_select_kernel(
    const unsigned long long* __restrict__ cand, const int* __restrict__ cnt,
    int* __restrict__ sel_idx, float* __restrict__ sel_score) {
  __shared__ unsigned long long key[CAP];   // 32 KB of the 320 KB WGP LDS
  const int c = blockIdx.x;
  const int n = min(cnt[c], CAP);

  // Bulk copy CAP entries (CAP*2 dwords) via the Tensor Data Mover.
  if (threadIdx.x == 0) {
    tdm_load_1d((unsigned)(size_t)&key[0], cand + (size_t)c * CAP, CAP * 2u);
  }
  __syncthreads();
  // Invalidate tail (slots never written this launch hold stale bytes).
  for (int i = threadIdx.x; i < CAP; i += 1024)
    if (i >= n) key[i] = 0ull;
  __syncthreads();

  // Bitonic sort, descending.
  for (unsigned k = 2; k <= CAP; k <<= 1) {
    for (unsigned j = k >> 1; j > 0; j >>= 1) {
      for (unsigned i = threadIdx.x; i < CAP; i += 1024) {
        unsigned ixj = i ^ j;
        if (ixj > i) {
          unsigned long long a = key[i], b = key[ixj];
          bool up = ((i & k) == 0);              // descending blocks
          bool sw = up ? (a < b) : (a > b);
          if (sw) { key[i] = b; key[ixj] = a; }
        }
      }
      __syncthreads();
    }
  }

  for (int j = threadIdx.x; j < TOPK; j += 1024) {
    unsigned long long kk = key[j];
    if (kk == 0ull) {
      sel_idx[c * TOPK + j] = -1;
      sel_score[c * TOPK + j] = -1.0f;
    } else {
      unsigned ub = (unsigned)(kk >> 32);
      sel_idx[c * TOPK + j] = (int)(kk & 0xFFFFFFFFu);
      sel_score[c * TOPK + j] = __uint_as_float(ub & 0x7FFFFFFFu);
    }
  }
}

// =====================================================================
// Stage C: gather deltas only for selected candidates (1.4 MB instead of
// 145 MB), decode + clip, greedy NMS entirely in LDS. One WG per class.
// =====================================================================
__global__ __launch_bounds__(1024) void decode_nms_kernel(
    const float* __restrict__ box_reg, const float* __restrict__ proposals,
    const int* __restrict__ sel_idx, const float* __restrict__ sel_score,
    float* __restrict__ fin_score, float* __restrict__ fin_box) {
  __shared__ float4 lbox[TOPK];
  __shared__ float  larea[TOPK];
  __shared__ float  lsc[TOPK];
  __shared__ int    lkeep[TOPK];
  const int c = blockIdx.x;           // foreground slot; class id = c+1

  for (int j = threadIdx.x; j < TOPK; j += 1024) {
    int i = sel_idx[c * TOPK + j];
    float s = sel_score[c * TOPK + j];
    float4 bb = make_float4(0.f, 0.f, 0.f, 0.f);
    if (i >= 0) {
      const float* pb = proposals + (size_t)i * 4;
      float w = pb[2] - pb[0] + 1.0f, hgt = pb[3] - pb[1] + 1.0f;
      float cx = pb[0] + 0.5f * w, cy = pb[1] + 0.5f * hgt;
      const float* dl = box_reg + (size_t)i * (NCLS * 4) + (size_t)(c + 1) * 4;
      float dx = dl[0] * 0.1f, dy = dl[1] * 0.1f;
      float dw = fminf(dl[2] * 0.2f, BBOX_CLIP);
      float dh = fminf(dl[3] * 0.2f, BBOX_CLIP);
      float pcx = dx * w + cx, pcy = dy * hgt + cy;
      float pw = __expf(dw) * w, ph = __expf(dh) * hgt;
      bb.x = fminf(fmaxf(pcx - 0.5f * pw, 0.f), kImgW - 1.0f);
      bb.y = fminf(fmaxf(pcy - 0.5f * ph, 0.f), kImgH - 1.0f);
      bb.z = fminf(fmaxf(pcx + 0.5f * pw - 1.0f, 0.f), kImgW - 1.0f);
      bb.w = fminf(fmaxf(pcy + 0.5f * ph - 1.0f, 0.f), kImgH - 1.0f);
    }
    lbox[j] = bb;
    larea[j] = (bb.z - bb.x + 1.0f) * (bb.w - bb.y + 1.0f);
    lsc[j] = s;
    lkeep[j] = (s > SCORE_THRESH) ? 1 : 0;
  }
  __syncthreads();

  // Greedy NMS: serial over i, parallel over j>i (thread j owns candidate j).
  const int j = (int)threadIdx.x;
  for (int i = 0; i < TOPK - 1; ++i) {
    if (j > i && j < TOPK && lkeep[i] && lkeep[j]) {
      float4 a = lbox[i], b = lbox[j];
      float xx1 = fmaxf(a.x, b.x), yy1 = fmaxf(a.y, b.y);
      float xx2 = fminf(a.z, b.z), yy2 = fminf(a.w, b.w);
      float iw = fmaxf(xx2 - xx1 + 1.0f, 0.f), ih = fmaxf(yy2 - yy1 + 1.0f, 0.f);
      float inter = iw * ih;
      float iou = inter / (larea[i] + larea[j] - inter);
      if (iou > NMS_THRESH) lkeep[j] = 0;
    }
    __syncthreads();
  }

  for (int t = threadIdx.x; t < TOPK; t += 1024) {
    fin_score[c * TOPK + t] = lkeep[t] ? lsc[t] : -1.0f;
    float4 bb = lbox[t];
    float* fb = &fin_box[(size_t)(c * TOPK + t) * 4];
    fb[0] = bb.x; fb[1] = bb.y; fb[2] = bb.z; fb[3] = bb.w;
  }
}

// =====================================================================
// Stage D: global top-100 over 90k scores via 100 argmax rounds.
// out layout: boxes[100][4] | scores[100] | labels[100]
// =====================================================================
__global__ __launch_bounds__(1024) void topk_out_kernel(
    float* __restrict__ fin_score, const float* __restrict__ fin_box,
    float* __restrict__ out) {
  __shared__ float rmax[1024];
  __shared__ int   ridx[1024];
  const int M = NFG * TOPK;
  for (int it = 0; it < NDETS; ++it) {
    float best = -1e30f; int bi = 0;
    for (int i = threadIdx.x; i < M; i += 1024) {
      float s = fin_score[i];
      if (s > best) { best = s; bi = i; }
    }
    rmax[threadIdx.x] = best; ridx[threadIdx.x] = bi;
    __syncthreads();
    for (int off = 512; off > 0; off >>= 1) {
      if ((int)threadIdx.x < off) {
        if (rmax[threadIdx.x + off] > rmax[threadIdx.x]) {
          rmax[threadIdx.x] = rmax[threadIdx.x + off];
          ridx[threadIdx.x] = ridx[threadIdx.x + off];
        }
      }
      __syncthreads();
    }
    if (threadIdx.x == 0) {
      int k = ridx[0];
      out[it * 4 + 0] = fin_box[(size_t)k * 4 + 0];
      out[it * 4 + 1] = fin_box[(size_t)k * 4 + 1];
      out[it * 4 + 2] = fin_box[(size_t)k * 4 + 2];
      out[it * 4 + 3] = fin_box[(size_t)k * 4 + 3];
      out[4 * NDETS + it] = rmax[0];
      out[5 * NDETS + it] = (float)(k / TOPK + 1);   // label
      fin_score[k] = -1e30f;                          // exclude for next round
    }
    __syncthreads();
  }
}

// =====================================================================
extern "C" void kernel_launch(void* const* d_in, const int* in_sizes, int n_in,
                              void* d_out, int out_size, void* d_ws, size_t ws_size,
                              hipStream_t stream) {
  (void)in_sizes; (void)n_in; (void)out_size; (void)ws_size;
  const float* logits    = (const float*)d_in[0];   // [100000, 91]
  const float* box_reg   = (const float*)d_in[1];   // [100000, 364]
  const float* proposals = (const float*)d_in[2];   // [100000, 4]
  float* out = (float*)d_out;                        // 600 floats

  // bump allocator over workspace (256B aligned)
  char* ws = (char*)d_ws;
  auto take = [&](size_t bytes) -> char* {
    char* p = ws;
    ws += (bytes + 255) & ~(size_t)255;
    return p;
  };
  float*              scores_t  = (float*)take((size_t)NFG * NPROP * sizeof(float)); // 36 MB
  int*                cnt       = (int*)take(NFG * sizeof(int));
  unsigned long long* cand      = (unsigned long long*)take((size_t)NFG * CAP * 8);
  int*                sel_idx   = (int*)take((size_t)NFG * TOPK * sizeof(int));
  float*              sel_score = (float*)take((size_t)NFG * TOPK * sizeof(float));
  float*              fin_score = (float*)take((size_t)NFG * TOPK * sizeof(float));
  float*              fin_box   = (float*)take((size_t)NFG * TOPK * 4 * sizeof(float));

  const int nwaves = (NPROP + 15) / 16;           // 6250 waves, 16 proposals each
  const int blocksA = (nwaves + 7) / 8;           // 8 waves / 256-thread block
  softmax_scores_kernel<<<blocksA, 256, 0, stream>>>(logits, scores_t);

  zero_cnt_kernel<<<1, 128, 0, stream>>>(cnt);

  dim3 gB1((NPROP + 255) / 256, NFG);
  compact_kernel<<<gB1, 256, 0, stream>>>(scores_t, cnt, cand);

  sort_select_kernel<<<NFG, 1024, 0, stream>>>(cand, cnt, sel_idx, sel_score);

  decode_nms_kernel<<<NFG, 1024, 0, stream>>>(box_reg, proposals, sel_idx, sel_score,
                                              fin_score, fin_box);

  topk_out_kernel<<<1, 1024, 0, stream>>>(fin_score, fin_box, out);
}
